// UCL_5832565588543
// MI455X (gfx1250) — matrix operand order repaired and checked by reading, not compiled, not run on hardware
//
#include <hip/hip_runtime.h>
#include <hip/hip_bf16.h>

// ---------------- types ----------------
typedef __attribute__((ext_vector_type(16))) _Float16 v16h;
typedef __attribute__((ext_vector_type(8)))  _Float16 v8h;
typedef __attribute__((ext_vector_type(8)))  float    v8f;
typedef __attribute__((ext_vector_type(4)))  unsigned int v4u;
typedef __attribute__((ext_vector_type(8)))  int      v8i;
typedef __attribute__((ext_vector_type(4)))  int      v4i;

union HalfVec { v16h v; v8h h[2]; };

#define NNODES 4096
#define INDIM  768
#define OUTDIM 512
#define INV_TEM 2.0f   // 1/0.5

// TDM availability (device pass only; host pass parses fallback path)
#if defined(__HIP_DEVICE_COMPILE__) && __has_builtin(__builtin_amdgcn_tensor_load_to_lds) && \
    __has_builtin(__builtin_amdgcn_s_wait_tensorcnt)
#define USE_TDM 1
#warning "CDNA5_PROBE: device pass USING tensor_load_to_lds (TDM) path"
#else
#define USE_TDM 0
#if defined(__HIP_DEVICE_COMPILE__)
#warning "CDNA5_PROBE: device pass FALLBACK (no tensor_load_to_lds builtin)"
#endif
#endif

__device__ inline v8f wmma_f16(v16h a, v16h b, v8f c) {
    return __builtin_amdgcn_wmma_f32_16x16x32_f16(
        /*neg_a=*/false, a, /*neg_b=*/false, b,
        /*c_mod=*/(short)0, c, /*reuse_a=*/false, /*reuse_b=*/false);
}

#if USE_TDM
// Issue a TDM load of a [16 x 512] f16 tile (rows contiguous, row stride 512 elems)
// from global memory into LDS at byte offset lds_off.  D# per CDNA5 ISA ch.8.
__device__ inline void tdm_load_tile(unsigned lds_off, const _Float16* gsrc) {
    unsigned long long ga = (unsigned long long)gsrc;
    v4u g0;
    g0.x = 1u;                                    // count=1, user descriptor
    g0.y = lds_off;                               // lds_addr [63:32]
    g0.z = (unsigned)(ga & 0xFFFFFFFFu);          // global_addr[31:0]
    g0.w = (unsigned)((ga >> 32) & 0x01FFFFFFu)   // global_addr[56:32]
         | (2u << 30);                            // type = 2 (image)
    v8i g1;
    g1[0] = (1 << 16);          // workgroup_mask=0, data_size=1 (2 bytes)
    g1[1] = (OUTDIM << 16);     // tensor_dim0[15:0] = 512 in bits 63:48
    g1[2] = (NNODES << 16);     // tensor_dim0 hi = 0; tensor_dim1[15:0] = 4096 in bits 95:80
    g1[3] = (OUTDIM << 16);     // tensor_dim1 hi = 0; tile_dim0 = 512 in bits 127:112
    g1[4] = 16;                 // tile_dim1 = 16; tile_dim2 = 0
    g1[5] = OUTDIM;             // tensor_dim0_stride = 512 (bits 207:160)
    g1[6] = 0;                  // stride hi, tensor_dim1_stride lo
    g1[7] = 0;
    v4i z4 = {0, 0, 0, 0};
#if __clang_major__ >= 23
    v8i z8 = {0, 0, 0, 0, 0, 0, 0, 0};
    __builtin_amdgcn_tensor_load_to_lds(g0, g1, z4, z4, z8, 0);
#else
    __builtin_amdgcn_tensor_load_to_lds(g0, g1, z4, z4, 0);
#endif
}
#endif

// ---------------- conversion kernels ----------------
__global__ void cvt_f32_to_f16(const float* __restrict__ in, _Float16* __restrict__ out, int n) {
    int i = blockIdx.x * 256 + threadIdx.x;
    if (i < n) out[i] = (_Float16)in[i];
}

// in: [gridDim.y][R][C] f32 row-major  ->  out: [gridDim.y][C][R] f16 (transposed)
__global__ void cvt_transpose_f16(const float* __restrict__ in, _Float16* __restrict__ out,
                                  int R, int C) {
    size_t base = (size_t)blockIdx.y * R * C;
    int i = blockIdx.x * 256 + threadIdx.x;
    if (i >= R * C) return;
    int r = i / C, c = i % C;
    out[base + (size_t)c * R + r] = (_Float16)in[base + i];
}

// ---------------- generic WMMA GEMM: C[M,N] = A[M,K](f16) * Bt[N,K](f16)^T + bias ----------------
// one 16x16 output tile per wave, 8 waves per block.
__global__ void gemm_wmma(const _Float16* __restrict__ A, const _Float16* __restrict__ Bt,
                          const float* __restrict__ bias, float* __restrict__ C,
                          int M, int N, int K) {
    int lane = threadIdx.x & 31;
    int wave = threadIdx.x >> 5;
    int tile = blockIdx.x * 8 + wave;
    int tilesN = N >> 4;
    int tm = tile / tilesN;
    int tn = tile - tm * tilesN;
    if (tm * 16 >= M) return;                 // wave-uniform guard

    int r0 = tm << 4, c0 = tn << 4;
    int m  = lane & 15;
    int kb = (lane >> 4) * 8;                 // A lane K-offset (chunks at kb and kb+16)
    int ko = (lane >> 4) * 16;                // unused; B loads the same kb, kb+16 chunks as A
    const _Float16* arow = A  + (size_t)(r0 + m) * K;
    const _Float16* brow = Bt + (size_t)(c0 + m) * K;

    v8f acc = {};
    for (int k0 = 0; k0 < K; k0 += 32) {
        __builtin_prefetch(arow + k0 + 256, 0, 3);   // -> global_prefetch_b8 (WGP scope)
        HalfVec a, b;
        a.h[0] = *(const v8h*)(arow + k0 + kb);
        a.h[1] = *(const v8h*)(arow + k0 + kb + 16);
        b.h[0] = *(const v8h*)(brow + k0 + kb);
        b.h[1] = *(const v8h*)(brow + k0 + kb + 16);
        acc = wmma_f16(a.v, b.v, acc);
    }
    // D layout: lanes 0-15 -> rows r0+v, col c0+lane ; lanes 16-31 -> rows r0+8+v
    int col   = c0 + (lane & 15);
    int rbase = r0 + ((lane >> 4) << 3);
    float bv  = bias[col];
    #pragma unroll
    for (int v = 0; v < 8; ++v)
        C[(size_t)(rbase + v) * N + col] = acc[v] + bv;
}

// ---------------- BatchNorm batch-stats (biased var) -> scale/shift ----------------
__global__ void bn_stats(const float* __restrict__ H, const float* __restrict__ gamma,
                         const float* __restrict__ beta,
                         float* __restrict__ scale, float* __restrict__ shift) {
    int c = blockIdx.x * 256 + threadIdx.x;
    if (c >= OUTDIM) return;
    float s = 0.f, sq = 0.f;
    for (int i = 0; i < NNODES; ++i) {
        float v = H[(size_t)i * OUTDIM + c];
        s += v; sq += v * v;
    }
    float mu  = s * (1.0f / NNODES);
    float var = sq * (1.0f / NNODES) - mu * mu;
    float inv = rsqrtf(var + 1e-5f);
    float sc  = gamma[c] * inv;
    scale[c] = sc;
    shift[c] = beta[c] - mu * sc;
}

__global__ void bn_relu_cvt(const float* __restrict__ H, const float* __restrict__ scale,
                            const float* __restrict__ shift, _Float16* __restrict__ out) {
    int i = blockIdx.x * 256 + threadIdx.x;  // over NNODES*OUTDIM
    int c = i & (OUTDIM - 1);
    float v = fmaxf(H[i] * scale[c] + shift[c], 0.0f);
    out[i] = (_Float16)v;
}

// ---------------- row L2 normalize + cvt to f16; one block per row ----------------
__global__ void l2norm_cvt(const float* __restrict__ O, _Float16* __restrict__ out) {
    __shared__ float red[256];
    int row = blockIdx.x;
    const float* r = O + (size_t)row * OUTDIM;
    float s = 0.f;
    for (int c = threadIdx.x; c < OUTDIM; c += 256) { float v = r[c]; s += v * v; }
    red[threadIdx.x] = s;
    __syncthreads();
    for (int st = 128; st > 0; st >>= 1) {
        if (threadIdx.x < st) red[threadIdx.x] += red[threadIdx.x + st];
        __syncthreads();
    }
    float inv = 1.0f / fmaxf(sqrtf(red[0]), 1e-12f);
    for (int c = threadIdx.x; c < OUTDIM; c += 256)
        out[(size_t)row * OUTDIM + c] = (_Float16)(r[c] * inv);
}

// ---------------- fused contrastive pair loss ----------------
// For anchor P and other Q (both [4096,512] f16, L2-normalized rows):
//   loss[i] = log( sum_j exp(2*p_i.p_j) + sum_j exp(2*p_i.q_j) - exp(2*p_i.p_i) )
//             - log( exp(2*p_i.q_i) )
// Each wave owns 16 rows (A-fragments cached in registers for all K=512).
// Column tiles of P and Q are staged in LDS: TDM double-buffered when available.
__global__ void pair_loss_kernel(const _Float16* __restrict__ P, const _Float16* __restrict__ Q,
                                 float* __restrict__ loss) {
    __shared__ _Float16 bufP[2][16 * OUTDIM];   // 2 x 16KB
    __shared__ _Float16 bufQ[2][16 * OUTDIM];   // 2 x 16KB  (total 64KB)

    int lane = threadIdx.x & 31;
    int wave = threadIdx.x >> 5;
    int I0 = blockIdx.x * 128 + wave * 16;
    int m  = lane & 15;
    int kb = (lane >> 4) * 8;
    int ko = (lane >> 4) * 16;

#if USE_TDM
    // prologue: start DMA of tile J=0 into buffer 0 while we fill the A-cache
    if (wave == 0) {
        tdm_load_tile((unsigned)(size_t)&bufP[0][0], P);
        tdm_load_tile((unsigned)(size_t)&bufQ[0][0], Q);
    }
#endif

    // cache this wave's 16 anchor rows: full K=512 in registers (reused 256x)
    HalfVec af[16];
    {
        const _Float16* arow = P + (size_t)(I0 + m) * OUTDIM;
        #pragma unroll
        for (int ks = 0; ks < 16; ++ks) {
            af[ks].h[0] = *(const v8h*)(arow + ks * 32 + kb);
            af[ks].h[1] = *(const v8h*)(arow + ks * 32 + kb + 16);
        }
    }

    float refl[8], betw[8], pos[8], dia[8];
    #pragma unroll
    for (int v = 0; v < 8; ++v) { refl[v] = 0.f; betw[v] = 0.f; pos[v] = 0.f; dia[v] = 0.f; }

    int rbase = I0 + ((lane >> 4) << 3);

    for (int J = 0; J < NNODES / 16; ++J) {
#if USE_TDM
        int cur = J & 1;
        if (wave == 0) {
            if (J + 1 < NNODES / 16) {
                // prefetch next tile into the other buffer, then wait for current
                tdm_load_tile((unsigned)(size_t)&bufP[cur ^ 1][0],
                              P + (size_t)(J + 1) * 16 * OUTDIM);
                tdm_load_tile((unsigned)(size_t)&bufQ[cur ^ 1][0],
                              Q + (size_t)(J + 1) * 16 * OUTDIM);
                __builtin_amdgcn_s_wait_tensorcnt(2);   // in-order: current pair done
            } else {
                __builtin_amdgcn_s_wait_tensorcnt(0);
            }
        }
        __syncthreads();                                 // tile J visible to all waves
        const _Float16* tileP = &bufP[cur][0];
        const _Float16* tileQ = &bufQ[cur][0];
#else
        __syncthreads();
        const v8h* srcP = (const v8h*)(P + (size_t)J * 16 * OUTDIM);
        const v8h* srcQ = (const v8h*)(Q + (size_t)J * 16 * OUTDIM);
        for (int t = threadIdx.x; t < 16 * OUTDIM / 8; t += 256) {
            ((v8h*)&bufP[0][0])[t] = srcP[t];
            ((v8h*)&bufQ[0][0])[t] = srcQ[t];
        }
        __syncthreads();
        const _Float16* tileP = &bufP[0][0];
        const _Float16* tileQ = &bufQ[0][0];
#endif

        v8f c1 = {}, c2 = {};
        #pragma unroll
        for (int ks = 0; ks < 16; ++ks) {
            const _Float16* bp = tileP + m * OUTDIM + ks * 32 + kb;
            const _Float16* bq = tileQ + m * OUTDIM + ks * 32 + kb;
            HalfVec b1, b2;
            b1.h[0] = *(const v8h*)bp; b1.h[1] = *(const v8h*)(bp + 16);
            b2.h[0] = *(const v8h*)bq; b2.h[1] = *(const v8h*)(bq + 16);
            c1 = wmma_f16(af[ks].v, b1.v, c1);
            c2 = wmma_f16(af[ks].v, b2.v, c2);
        }

        int col = J * 16 + (lane & 15);
        #pragma unroll
        for (int v = 0; v < 8; ++v) {
            float e1 = __expf(c1[v] * INV_TEM);
            float e2 = __expf(c2[v] * INV_TEM);
            refl[v] += e1;
            betw[v] += e2;
            if (rbase + v == col) { pos[v] += e2; dia[v] += e1; }
        }
#if USE_TDM
        __syncthreads();   // everyone done reading buf[cur] before wave0 reissues into it
#endif
    }

    // reduce across the 16 lanes of each half (rows 0-7 live in lanes 0-15, rows 8-15 in 16-31)
    #pragma unroll
    for (int mask = 8; mask > 0; mask >>= 1) {
        #pragma unroll
        for (int v = 0; v < 8; ++v) {
            refl[v] += __shfl_xor(refl[v], mask, 32);
            betw[v] += __shfl_xor(betw[v], mask, 32);
            pos[v]  += __shfl_xor(pos[v],  mask, 32);
            dia[v]  += __shfl_xor(dia[v],  mask, 32);
        }
    }
    if ((lane & 15) == 0) {
        #pragma unroll
        for (int v = 0; v < 8; ++v) {
            float denom = refl[v] + betw[v] - dia[v];
            loss[rbase + v] = __logf(denom) - __logf(pos[v]);
        }
    }
}

// ---------------- final fixed-order reduction ----------------
__global__ void final_reduce(const float* __restrict__ loss, float* __restrict__ out) {
    __shared__ float red[256];
    float s = 0.f;
    for (int i = threadIdx.x; i < 6 * NNODES; i += 256) s += loss[i];
    red[threadIdx.x] = s;
    __syncthreads();
    for (int st = 128; st > 0; st >>= 1) {
        if (threadIdx.x < st) red[threadIdx.x] += red[threadIdx.x + st];
        __syncthreads();
    }
    if (threadIdx.x == 0) out[0] = red[0] * (0.5f / (3.0f * NNODES));
}

// ---------------- host launch ----------------
extern "C" void kernel_launch(void* const* d_in, const int* in_sizes, int n_in,
                              void* d_out, int out_size, void* d_ws, size_t ws_size,
                              hipStream_t stream) {
    const float* doc[3] = { (const float*)d_in[0], (const float*)d_in[1], (const float*)d_in[2] };
    const float* W1s    = (const float*)d_in[3];
    const float* b1s    = (const float*)d_in[4];
    const float* gammas = (const float*)d_in[5];
    const float* betas  = (const float*)d_in[6];
    const float* W2s    = (const float*)d_in[7];
    const float* b2s    = (const float*)d_in[8];

    // workspace bump allocator (256B aligned)
    char* ws = (char*)d_ws;
    size_t off = 0;
    auto alloc = [&](size_t bytes) {
        void* p = ws + off;
        off += (bytes + 255) & ~(size_t)255;
        return p;
    };
    _Float16* Xbf  = (_Float16*)alloc((size_t)3 * NNODES * INDIM * 2);
    _Float16* W1t  = (_Float16*)alloc((size_t)6 * OUTDIM * INDIM * 2);   // [6][512][768]
    _Float16* W2t  = (_Float16*)alloc((size_t)6 * OUTDIM * OUTDIM * 2);  // [6][512][512]
    _Float16* Abf  = (_Float16*)alloc((size_t)6 * NNODES * OUTDIM * 2);
    float*    H    = (float*)   alloc((size_t)NNODES * OUTDIM * 4);
    float*    O    = (float*)   alloc((size_t)NNODES * OUTDIM * 4);
    _Float16* Hbf  = (_Float16*)alloc((size_t)NNODES * OUTDIM * 2);
    float*    scl  = (float*)   alloc(OUTDIM * 4);
    float*    shf  = (float*)   alloc(OUTDIM * 4);
    float*    lossb= (float*)   alloc((size_t)6 * NNODES * 4);
    (void)ws_size; (void)n_in; (void)in_sizes; (void)out_size;

    // 1. convert docs to f16
    for (int d = 0; d < 3; ++d)
        cvt_f32_to_f16<<<(NNODES * INDIM + 255) / 256, 256, 0, stream>>>(
            doc[d], Xbf + (size_t)d * NNODES * INDIM, NNODES * INDIM);

    // 2. transpose+convert weights: W1t[k][n][kk] = W1s[k][kk][n]
    cvt_transpose_f16<<<dim3((INDIM * OUTDIM + 255) / 256, 6), 256, 0, stream>>>(
        W1s, W1t, INDIM, OUTDIM);
    cvt_transpose_f16<<<dim3((OUTDIM * OUTDIM + 255) / 256, 6), 256, 0, stream>>>(
        W2s, W2t, OUTDIM, OUTDIM);

    // 3. six projectors (k uses doc src[k] and parameter set k)
    const int src[6] = {0, 1, 0, 2, 1, 2};
    const int gemmBlocks = (NNODES / 16) * (OUTDIM / 16) / 8;  // 1024
    for (int k = 0; k < 6; ++k) {
        gemm_wmma<<<gemmBlocks, 256, 0, stream>>>(
            Xbf + (size_t)src[k] * NNODES * INDIM, W1t + (size_t)k * OUTDIM * INDIM,
            b1s + k * OUTDIM, H, NNODES, OUTDIM, INDIM);
        bn_stats<<<2, 256, 0, stream>>>(H, gammas + k * OUTDIM, betas + k * OUTDIM, scl, shf);
        bn_relu_cvt<<<(NNODES * OUTDIM) / 256, 256, 0, stream>>>(H, scl, shf, Hbf);
        gemm_wmma<<<gemmBlocks, 256, 0, stream>>>(
            Hbf, W2t + (size_t)k * OUTDIM * OUTDIM,
            b2s + k * OUTDIM, O, NNODES, OUTDIM, OUTDIM);
        l2norm_cvt<<<NNODES, 256, 0, stream>>>(O, Abf + (size_t)k * NNODES * OUTDIM);
    }

    // 4. six loss directions: pairs (o0,o1),(o2,o3),(o4,o5) both ways
    const int anc[6] = {0, 1, 2, 3, 4, 5};
    const int oth[6] = {1, 0, 3, 2, 5, 4};
    for (int d = 0; d < 6; ++d)
        pair_loss_kernel<<<NNODES / 128, 256, 0, stream>>>(
            Abf + (size_t)anc[d] * NNODES * OUTDIM,
            Abf + (size_t)oth[d] * NNODES * OUTDIM,
            lossb + (size_t)d * NNODES);

    // 5. deterministic final reduction
    final_reduce<<<1, 256, 0, stream>>>(lossb, (float*)d_out);
}
